// MultiEmbodimentActionEncoder_24962349924928
// MI455X (gfx1250) — compile-verified, matrix operand order checked by reading, not run on hardware
//
#include <hip/hip_runtime.h>

// MI455X / gfx1250, wave32. bf16 WMMA (V_WMMA_F32_16X16X32_BF16), f32 accumulate.
// Weights pre-converted once per launch into WMMA-operand-tiled bf16 in d_ws.

typedef __attribute__((ext_vector_type(16))) __bf16         v16bf;
typedef __attribute__((ext_vector_type(16))) unsigned short v16u;
typedef __attribute__((ext_vector_type(8)))  unsigned short v8u;
typedef __attribute__((ext_vector_type(8)))  unsigned int   v8ui;
typedef __attribute__((ext_vector_type(8)))  float          v8f;

#define T_DIM   64
#define A_DIM   32
#define H_DIM   512
#define LDA_ACT 40     // 32 + 8 bf16 pad (rows 16B-aligned)
#define LDA     520    // 512 + 8 bf16 pad

// pack two f32 -> packed bf16 pair {hi,lo} with a single V_PERM_B32 (truncating)
__device__ __forceinline__ unsigned int pack2bf(float flo, float fhi) {
  return __builtin_amdgcn_perm(__float_as_uint(fhi), __float_as_uint(flo), 0x07060302u);
}
__device__ __forceinline__ unsigned short bftrunc(float f) {
  return (unsigned short)(__float_as_uint(f) >> 16);
}

// ---------------------------------------------------------------------------
// Weight pre-conversion: f32 row-major (K x 512) -> bf16 WMMA B-operand tiles.
// Tile (kt, nt): 32 lanes x 16 bf16 (32 contiguous bytes per lane).
// Lane L: n = nt*16 + (L&15), kofs = kt*32 + (L>>4)*8;
// element pairs j<4: K = kofs+2j, kofs+2j+1 ; j>=4: K = kofs+16+2(j-4), +1.
// ---------------------------------------------------------------------------
__global__ __launch_bounds__(256)
void convert_weights_kernel(const float* __restrict__ W, unsigned int* __restrict__ dst,
                            int nkt, int catStride /* f32 elems per cat */) {
  const int slot = blockIdx.x * 256 + threadIdx.x;   // (cat, kt, nt, lane)
  const int lane = slot & 31;
  const int nt   = (slot >> 5) & 31;
  const int ktc  = slot >> 10;
  const int kt   = ktc % nkt;
  const int cat  = ktc / nkt;
  const float* Wb = W + (size_t)cat * catStride;
  const int n    = nt * 16 + (lane & 15);
  const int kofs = kt * 32 + ((lane >> 4) & 1) * 8;
  const float* wp = Wb + (size_t)kofs * H_DIM + n;
  v8ui p;
#pragma unroll
  for (int j = 0; j < 4; ++j)
    p[j]     = pack2bf(wp[(size_t)(2*j)      * H_DIM], wp[(size_t)(2*j + 1)  * H_DIM]);
#pragma unroll
  for (int j = 0; j < 4; ++j)
    p[4 + j] = pack2bf(wp[(size_t)(16 + 2*j) * H_DIM], wp[(size_t)(17 + 2*j) * H_DIM]);
  *(v8ui*)(dst + (size_t)slot * 8) = p;
}

// A operand (16x32) from bf16 LDS: two 16B chunks at c0 and c0+16 (ds_load_b128 x2).
__device__ __forceinline__ v16bf load_a_tile(const unsigned short* A, int ld, int row, int c0) {
  const unsigned short* ap = A + row * ld + c0;
  v8u lo = *(const v8u*)ap;
  v8u hi = *(const v8u*)(ap + 16);
  v16u t = __builtin_shufflevector(lo, hi, 0,1,2,3,4,5,6,7,8,9,10,11,12,13,14,15);
  return __builtin_bit_cast(v16bf, t);
}

// B operand: preconverted path = one 32B contiguous load per lane.
__device__ __forceinline__ v16bf load_b_pre(const unsigned short* Wcv, int kt, int nt, int lane) {
  const v16u* tp = (const v16u*)(Wcv + ((((size_t)kt * 32 + nt) * 32) + lane) * 16);
  return __builtin_bit_cast(v16bf, *tp);
}
// B operand: fallback = 16 strided f32 loads + 8 V_PERM packs.
__device__ __forceinline__ v16bf load_b_f32(const float* __restrict__ W, int kofs, int n) {
  const float* wp = W + (size_t)kofs * H_DIM + n;
  v8ui p;
#pragma unroll
  for (int j = 0; j < 4; ++j)
    p[j]     = pack2bf(wp[(size_t)(2*j)      * H_DIM], wp[(size_t)(2*j + 1)  * H_DIM]);
#pragma unroll
  for (int j = 0; j < 4; ++j)
    p[4 + j] = pack2bf(wp[(size_t)(16 + 2*j) * H_DIM], wp[(size_t)(17 + 2*j) * H_DIM]);
  return __builtin_bit_cast(v16bf, p);
}

template <bool PRE>
__global__ __launch_bounds__(256, 1)
void mea_encoder_kernel(const float* __restrict__ actions,
                        const float* __restrict__ timesteps,
                        const int*   __restrict__ cat_ids,
                        const float* __restrict__ W1, const float* __restrict__ b1,
                        const float* __restrict__ W2, const float* __restrict__ b2,
                        const float* __restrict__ W3, const float* __restrict__ b3,
                        const unsigned short* __restrict__ W1cv,
                        const unsigned short* __restrict__ W2cv,
                        const unsigned short* __restrict__ W3cv,
                        float* __restrict__ out)
{
  __shared__ __attribute__((aligned(16))) unsigned short ldsAct[T_DIM * LDA_ACT];
  __shared__ __attribute__((aligned(16))) unsigned short ldsA  [T_DIM * LDA];
  __shared__ __attribute__((aligned(16))) unsigned short ldsHid[T_DIM * LDA];
  __shared__ float ldsTau[H_DIM];
  __shared__ float ldsC  [H_DIM];

  const int b      = blockIdx.x;
  const int tid    = threadIdx.x;
  const int lane   = tid & 31;
  const int wave   = tid >> 5;
  const int lanelo = lane & 15;
  const int lanehi = (lane >> 4) & 1;
  const int cat    = cat_ids[b];

  const float* W1b = W1 + (size_t)cat * A_DIM * H_DIM;
  const float* W2a = W2 + (size_t)cat * (2 * H_DIM) * H_DIM;
  const float* W2t = W2a + (size_t)H_DIM * H_DIM;              // tau half, stays f32
  const float* W3b = W3 + (size_t)cat * H_DIM * H_DIM;
  const float* b1b = b1 + (size_t)cat * H_DIM;
  const float* b2b = b2 + (size_t)cat * H_DIM;
  const float* b3b = b3 + (size_t)cat * H_DIM;
  const unsigned short* W1cvb = W1cv + (size_t)cat * (1  * 32 * 32 * 16);
  const unsigned short* W2cvb = W2cv + (size_t)cat * (16 * 32 * 32 * 16);
  const unsigned short* W3cvb = W3cv + (size_t)cat * (16 * 32 * 32 * 16);

  // ---- phase 0: stage actions (bf16) + sinusoidal tau ----
  {
    const float* act = actions + (size_t)b * T_DIM * A_DIM;
    for (int i = tid; i < T_DIM * A_DIM; i += 256) {
      int r = i >> 5, c = i & 31;
      ldsAct[r * LDA_ACT + c] = bftrunc(act[i]);
    }
    float ts  = timesteps[b];
    float fr  = __expf(-9.210340371976184f * (1.0f / 256.0f) * (float)tid);
    float ang = ts * fr;
    ldsTau[tid]       = __sinf(ang);
    ldsTau[tid + 256] = __cosf(ang);
  }
  __syncthreads();

  // ---- layer 1: a_emb = actions @ W1 + b1 -> ldsA (bf16) ----
#pragma unroll 1
  for (int i = 0; i < 4; ++i) {
    const int nt = wave * 4 + i;
    const int n  = nt * 16 + lanelo;
    v16bf bm = PRE ? load_b_pre(W1cvb, 0, nt, lane)
                   : load_b_f32(W1b, lanehi * 8, n);
    const float bias = b1b[n];
#pragma unroll
    for (int mt = 0; mt < 4; ++mt) {
      v16bf am = load_a_tile(ldsAct, LDA_ACT, mt * 16 + lanelo, lanehi * 8);
      v8f acc = {};
      acc = __builtin_amdgcn_wmma_f32_16x16x32_bf16(false, am, false, bm,
                                                    (short)0, acc, false, false);
#pragma unroll
      for (int r = 0; r < 8; ++r)
        ldsA[(mt * 16 + r + lanehi * 8) * LDA + n] = bftrunc(acc[r] + bias);
    }
  }

  // ---- c = tau @ W2[H:,:] + b2 (per-batch 512-vector, VALU GEMV) ----
  {
    float c0 = b2b[tid];
    float c1 = b2b[tid + 256];
#pragma unroll 4
    for (int k = 0; k < H_DIM; ++k) {
      float tv = ldsTau[k];
      const float* wr = W2t + (size_t)k * H_DIM;
      c0 = fmaf(tv, wr[tid],       c0);
      c1 = fmaf(tv, wr[tid + 256], c1);
    }
    ldsC[tid]       = c0;
    ldsC[tid + 256] = c1;
  }
  __syncthreads();

  // ---- layer 2: hidden = swish(a_emb @ W2a + c) -> ldsHid (bf16) ----
#pragma unroll 1
  for (int i = 0; i < 4; ++i) {
    const int nt = wave * 4 + i;
    const int n  = nt * 16 + lanelo;
    v8f acc[4] = {};
    for (int kt = 0; kt < 16; ++kt) {
      v16bf bm = PRE ? load_b_pre(W2cvb, kt, nt, lane)
                     : load_b_f32(W2a, kt * 32 + lanehi * 8, n);
      if (PRE && kt < 15)
        __builtin_prefetch(W2cvb + ((((size_t)(kt + 1) * 32 + nt) * 32) + lane) * 16, 0, 0);
#pragma unroll
      for (int mt = 0; mt < 4; ++mt) {
        v16bf am = load_a_tile(ldsA, LDA, mt * 16 + lanelo, kt * 32 + lanehi * 8);
        acc[mt] = __builtin_amdgcn_wmma_f32_16x16x32_bf16(false, am, false, bm,
                                                          (short)0, acc[mt], false, false);
      }
    }
    const float cj = ldsC[n];
#pragma unroll
    for (int mt = 0; mt < 4; ++mt) {
#pragma unroll
      for (int r = 0; r < 8; ++r) {
        float v = acc[mt][r] + cj;
        v = v / (1.0f + __expf(-v));                       // swish
        ldsHid[(mt * 16 + r + lanehi * 8) * LDA + n] = bftrunc(v);
      }
    }
  }
  __syncthreads();

  // ---- layer 3: out = hidden @ W3 + b3 -> global f32 ----
  float* outb = out + (size_t)b * T_DIM * H_DIM;
#pragma unroll 1
  for (int i = 0; i < 4; ++i) {
    const int nt = wave * 4 + i;
    const int n  = nt * 16 + lanelo;
    v8f acc[4] = {};
    for (int kt = 0; kt < 16; ++kt) {
      v16bf bm = PRE ? load_b_pre(W3cvb, kt, nt, lane)
                     : load_b_f32(W3b, kt * 32 + lanehi * 8, n);
      if (PRE && kt < 15)
        __builtin_prefetch(W3cvb + ((((size_t)(kt + 1) * 32 + nt) * 32) + lane) * 16, 0, 0);
#pragma unroll
      for (int mt = 0; mt < 4; ++mt) {
        v16bf am = load_a_tile(ldsHid, LDA, mt * 16 + lanelo, kt * 32 + lanehi * 8);
        acc[mt] = __builtin_amdgcn_wmma_f32_16x16x32_bf16(false, am, false, bm,
                                                          (short)0, acc[mt], false, false);
      }
    }
    const float bias = b3b[n];
#pragma unroll
    for (int mt = 0; mt < 4; ++mt) {
#pragma unroll
      for (int r = 0; r < 8; ++r)
        outb[(size_t)(mt * 16 + r + lanehi * 8) * H_DIM + n] = acc[mt][r] + bias;
    }
  }
}

extern "C" void kernel_launch(void* const* d_in, const int* in_sizes, int n_in,
                              void* d_out, int out_size, void* d_ws, size_t ws_size,
                              hipStream_t stream) {
  (void)n_in; (void)out_size;
  const float* actions   = (const float*)d_in[0];
  const float* timesteps = (const float*)d_in[1];
  const int*   cat_ids   = (const int*)d_in[2];
  const float* W1 = (const float*)d_in[3];
  const float* b1 = (const float*)d_in[4];
  const float* W2 = (const float*)d_in[5];
  const float* b2 = (const float*)d_in[6];
  const float* W3 = (const float*)d_in[7];
  const float* b3 = (const float*)d_in[8];
  float* out = (float*)d_out;

  const int B = in_sizes[1];                        // batches
  const int E = in_sizes[3] / (A_DIM * H_DIM);      // embodiments

  const size_t w1cvBytes = (size_t)E * 1  * 1024 * 16 * 2;   // E * 32KB
  const size_t w2cvBytes = (size_t)E * 16 * 1024 * 16 * 2;   // E * 512KB
  const size_t w3cvBytes = w2cvBytes;
  const size_t needed    = w1cvBytes + w2cvBytes + w3cvBytes;

  dim3 block(256), grid(B);
  if (d_ws != nullptr && ws_size >= needed) {
    unsigned short* W1cv = (unsigned short*)d_ws;
    unsigned short* W2cv = (unsigned short*)((char*)d_ws + w1cvBytes);
    unsigned short* W3cv = (unsigned short*)((char*)d_ws + w1cvBytes + w2cvBytes);
    // one thread per (cat, kt, nt, lane) slot
    hipLaunchKernelGGL(convert_weights_kernel, dim3((E * 1  * 1024) / 256), block, 0, stream,
                       W1, (unsigned int*)W1cv, 1,  A_DIM * H_DIM);
    hipLaunchKernelGGL(convert_weights_kernel, dim3((E * 16 * 1024) / 256), block, 0, stream,
                       W2, (unsigned int*)W2cv, 16, 2 * H_DIM * H_DIM);   // rows 0..511 only
    hipLaunchKernelGGL(convert_weights_kernel, dim3((E * 16 * 1024) / 256), block, 0, stream,
                       W3, (unsigned int*)W3cv, 16, H_DIM * H_DIM);
    hipLaunchKernelGGL((mea_encoder_kernel<true>), grid, block, 0, stream,
                       actions, timesteps, cat_ids, W1, b1, W2, b2, W3, b3,
                       W1cv, W2cv, W3cv, out);
  } else {
    hipLaunchKernelGGL((mea_encoder_kernel<false>), grid, block, 0, stream,
                       actions, timesteps, cat_ids, W1, b1, W2, b2, W3, b3,
                       nullptr, nullptr, nullptr, out);
  }
}
